// Model_85873576116428
// MI455X (gfx1250) — compile-verified, matrix operand order checked by reading, not run on hardware
//
#include <hip/hip_runtime.h>
#include <hip/hip_bf16.h>

typedef __attribute__((ext_vector_type(16))) _Float16 v16h;
typedef __attribute__((ext_vector_type(8)))  float    v8f;

constexpr int NN   = 4096;   // nodes
constexpr int TT   = 16;     // timesteps (= out cols of final layer)
constexpr int FFE  = 4;      // node features
constexpr int NHD  = 4;      // heads

// ---------------- workspace layout (bytes, all 64B aligned) ----------------
constexpr size_t OFF_H     = 0;                                   // h   [N,64] f32
constexpr size_t OFF_WH    = OFF_H     + (size_t)NN * 64  * 4;    // Wh  [N,256] f32
constexpr size_t OFF_S1    = OFF_WH    + (size_t)NN * 256 * 4;    // s1  [4][N]
constexpr size_t OFF_S2    = OFF_S1    + (size_t)4 * NN * 4;      // s2  [4][N]
constexpr size_t OFF_TOPF  = OFF_S2    + (size_t)4 * NN * 4;      // top2 floats (max,2nd)/head
constexpr size_t OFF_TOPI  = OFF_TOPF  + 64;                      // argmax ints
constexpr size_t OFF_M     = OFF_TOPI  + 64;                      // m   [4][N]
constexpr size_t OFF_Z     = OFF_M     + (size_t)4 * NN * 4;      // Z   [4][N]
constexpr size_t OFF_BH    = OFF_Z     + (size_t)4 * NN * 4;      // Wh f16 B-layout blobs
constexpr size_t OFF_HCAT  = OFF_BH    + (size_t)16 * 128 * 512 * 2; // hcat [N,256] f32
constexpr size_t OFF_WHO   = OFF_HCAT  + (size_t)NN * 256 * 4;    // Wh_out [N,16] f32
constexpr size_t OFF_S1O   = OFF_WHO   + (size_t)NN * 16 * 4;
constexpr size_t OFF_S2O   = OFF_S1O   + (size_t)NN * 4;
constexpr size_t OFF_TOPFO = OFF_S2O   + (size_t)NN * 4;
constexpr size_t OFF_TOPIO = OFF_TOPFO + 64;
constexpr size_t OFF_MO    = OFF_TOPIO + 64;
constexpr size_t OFF_ZO    = OFF_MO    + (size_t)NN * 4;
constexpr size_t OFF_BO    = OFF_ZO    + (size_t)NN * 4;          // Wh_out f16 blob (128KB)

// w(u) = exp(leaky(u) - m), branch-free
__device__ __forceinline__ float expw(float u, float mi) {
  float lk = u >= 0.f ? u : 0.01f * u;     // v_cndmask, no control flow
  return __expf(lk - mi);
}

// ---------------- h[n, t*4+f] = x[0,t,n,f] ----------------
__global__ __launch_bounds__(256) void k_prep_h(const float* __restrict__ x,
                                                float* __restrict__ h32) {
  int idx = blockIdx.x * 256 + threadIdx.x;
  if (idx >= NN * 64) return;
  int n = idx >> 6, k = idx & 63;
  int t = k >> 2, f = k & 3;
  h32[idx] = x[((size_t)t * NN + n) * FFE + f];
}

// ---------------- Wh[n, hd*64+c] = sum_k h[n,k] * W[hd][k][c]  (f32, tiny GEMM) ----
__global__ __launch_bounds__(256) void k_gemm_wh(const float* __restrict__ h32,
                                                 const float* __restrict__ W,
                                                 float* __restrict__ Wh) {
  int idx = blockIdx.x * 256 + threadIdx.x;
  if (idx >= NN * 256) return;
  int n = idx >> 8, hc = idx & 255;
  int hd = hc >> 6, c = hc & 63;
  const float* hr = h32 + (size_t)n * 64;
  const float* wc = W + (size_t)hd * 64 * 64 + c;
  float acc = 0.f;
#pragma unroll 16
  for (int k = 0; k < 64; ++k) acc += hr[k] * wc[k * 64];
  Wh[idx] = acc;
}

// ---------------- s1[hd][n] = Wh_row . a1 ;  s2 likewise ----------------
__global__ __launch_bounds__(256) void k_s12(const float* __restrict__ Wh,
                                             const float* __restrict__ a1,
                                             const float* __restrict__ a2,
                                             float* __restrict__ s1,
                                             float* __restrict__ s2) {
  int idx = blockIdx.x * 256 + threadIdx.x;
  if (idx >= NHD * NN) return;
  int hd = idx >> 12, n = idx & (NN - 1);
  const float* wr = Wh + (size_t)n * 256 + hd * 64;
  const float* A1 = a1 + hd * 64;
  const float* A2 = a2 + hd * 64;
  float d1 = 0.f, d2 = 0.f;
#pragma unroll 16
  for (int c = 0; c < 64; ++c) { float v = wr[c]; d1 += v * A1[c]; d2 += v * A2[c]; }
  s1[idx] = d1; s2[idx] = d2;
}

// ---------------- top-2 (max, second, argmax) of s2 per head ----------------
__global__ __launch_bounds__(256) void k_top2(const float* __restrict__ s2,
                                              float* __restrict__ outf,
                                              int* __restrict__ outi) {
  __shared__ float smx[256], smx2[256];
  __shared__ int   sarg[256];
  int hd = blockIdx.x;
  const float* v = s2 + (size_t)hd * NN;
  float mx = -INFINITY, mx2 = -INFINITY; int arg = -1;
  for (int j = threadIdx.x; j < NN; j += 256) {
    float t = v[j];
    if (t > mx)      { mx2 = mx; mx = t; arg = j; }
    else if (t > mx2) mx2 = t;
  }
  int tid = threadIdx.x;
  smx[tid] = mx; smx2[tid] = mx2; sarg[tid] = arg;
  __syncthreads();
  for (int s = 128; s > 0; s >>= 1) {
    if (tid < s) {
      float omx = smx[tid + s], omx2 = smx2[tid + s];
      if (omx > smx[tid]) {
        smx2[tid] = fmaxf(smx[tid], omx2);
        smx[tid] = omx; sarg[tid] = sarg[tid + s];
      } else {
        smx2[tid] = fmaxf(smx2[tid], omx);
      }
    }
    __syncthreads();
  }
  if (tid == 0) { outf[hd * 2] = smx[0]; outf[hd * 2 + 1] = smx2[0]; outi[hd] = sarg[0]; }
}

// ------- pass A: exact row max m_i (O(1) via top-2) and Z_i (branch-free) -------
__global__ __launch_bounds__(256) void k_passA(const float* __restrict__ s1,
                                               const float* __restrict__ s2,
                                               const float* __restrict__ topf,
                                               const int* __restrict__ topi,
                                               float* __restrict__ m,
                                               float* __restrict__ Z) {
  __shared__ __align__(16) float s2s[NN];
  int hd = blockIdx.y;
  {
    const float4* src = (const float4*)(s2 + (size_t)hd * NN);
    float4* dst = (float4*)s2s;
    for (int j = threadIdx.x; j < NN / 4; j += 256) dst[j] = src[j];
  }
  __syncthreads();
  int i = blockIdx.x * 256 + threadIdx.x;
  float s1i = s1[(size_t)hd * NN + i];
  float s2ex = (topi[hd] == i) ? topf[hd * 2 + 1] : topf[hd * 2];
  float t = s1i + s2ex;
  float mi = t >= 0.f ? t : 0.01f * t;     // leaky monotone => exact row max
  float z0 = 0.f, z1 = 0.f, z2 = 0.f, z3 = 0.f;
  for (int j = 0; j < NN; j += 4) {        // full sum, diagonal removed below
    float4 q = *(const float4*)(s2s + j);
    z0 += expw(s1i + q.x, mi);
    z1 += expw(s1i + q.y, mi);
    z2 += expw(s1i + q.z, mi);
    z3 += expw(s1i + q.w, mi);
  }
  float z = (z0 + z1) + (z2 + z3) - expw(s1i + s2s[i], mi);
  m[(size_t)hd * NN + i] = mi;
  Z[(size_t)hd * NN + i] = z;
}

// ------- pack f32 matrix [N, ncols] into WMMA f16 B-operand blobs -------
// blob[(((g*128)+c)*32 + lane)*16 + v] = M[c*32 + K, g*16 + (lane&15)]
// K = v + (v>=8 ? 8:0) + (lane>=16 ? 8:0)    (ISA 16-bit B layout)
__global__ __launch_bounds__(256) void k_packB(const float* __restrict__ M,
                                               _Float16* __restrict__ blob,
                                               int ngroups, int ncols) {
  int idx = blockIdx.x * 256 + threadIdx.x;
  if (idx >= ngroups * 128 * 512) return;
  int v = idx & 15, lane = (idx >> 4) & 31, c = (idx >> 9) & 127, g = idx >> 16;
  int K = (v < 8 ? v : v + 8) + (lane < 16 ? 0 : 8);
  int j = c * 32 + K;
  int col = g * 16 + (lane & 15);
  blob[idx] = (_Float16)M[(size_t)j * ncols + col];
}

// ------- pass B: out[i,:] = (sum_ALL_j w_ij*Wh[j,:] - w_ii*Wh[i,:]) / Z_i -------
// One wave = 16 rows x (NT*16) cols. A built in-register branch-free, B pre-packed
// f16, f32 accumulate via v_wmma_f32_16x16x32_f16. Diagonal fixed in the epilogue
// with the same f16 roundings the WMMA saw, so it cancels exactly.
template <int NT, bool DO_ELU>
__global__ __launch_bounds__(256) void k_passB(const float* __restrict__ s1,
                                               const float* __restrict__ s2,
                                               const float* __restrict__ m,
                                               const float* __restrict__ Z,
                                               const _Float16* __restrict__ blob,
                                               const float* __restrict__ Wsrc,
                                               float* __restrict__ outp,
                                               int out_stride) {
  __shared__ __align__(16) float s2s[NN];
  int hd = blockIdx.y;
  {
    const float4* src = (const float4*)(s2 + (size_t)hd * NN);
    float4* dst = (float4*)s2s;
    for (int j = threadIdx.x; j < NN / 4; j += 256) dst[j] = src[j];
  }
  __syncthreads();

  int wave = threadIdx.x >> 5;
  int lane = threadIdx.x & 31;
  int hi   = lane >> 4;                     // lane half selects K sub-block
  int ibase = (blockIdx.x * 8 + wave) * 16;
  int i = ibase + (lane & 15);              // A-matrix row owned by this lane
  float s1i = s1[(size_t)hd * NN + i];
  float mi  = m [(size_t)hd * NN + i];

  v8f acc[NT] = {};
  const v16h* B = (const v16h*)blob + (size_t)hd * NT * 128 * 32;

  for (int c = 0; c < 128; ++c) {          // 128 chunks of 32 j's
    int j0 = c * 32 + hi * 8;
    // per-lane K runs: j0..j0+7 and j0+16..j0+23, both 32B aligned -> 4x ds_load_b128
    float4 q0 = *(const float4*)(s2s + j0);
    float4 q1 = *(const float4*)(s2s + j0 + 4);
    float4 q2 = *(const float4*)(s2s + j0 + 16);
    float4 q3 = *(const float4*)(s2s + j0 + 20);
    v16h a;
    a[0]  = (_Float16)expw(s1i + q0.x, mi);
    a[1]  = (_Float16)expw(s1i + q0.y, mi);
    a[2]  = (_Float16)expw(s1i + q0.z, mi);
    a[3]  = (_Float16)expw(s1i + q0.w, mi);
    a[4]  = (_Float16)expw(s1i + q1.x, mi);
    a[5]  = (_Float16)expw(s1i + q1.y, mi);
    a[6]  = (_Float16)expw(s1i + q1.z, mi);
    a[7]  = (_Float16)expw(s1i + q1.w, mi);
    a[8]  = (_Float16)expw(s1i + q2.x, mi);
    a[9]  = (_Float16)expw(s1i + q2.y, mi);
    a[10] = (_Float16)expw(s1i + q2.z, mi);
    a[11] = (_Float16)expw(s1i + q2.w, mi);
    a[12] = (_Float16)expw(s1i + q3.x, mi);
    a[13] = (_Float16)expw(s1i + q3.y, mi);
    a[14] = (_Float16)expw(s1i + q3.z, mi);
    a[15] = (_Float16)expw(s1i + q3.w, mi);
    if (c + 1 < 128) __builtin_prefetch(&B[(size_t)(c + 1) * 32 + lane]);
#pragma unroll
    for (int t = 0; t < NT; ++t) {
      v16h b = B[((size_t)t * 128 + c) * 32 + lane];
      acc[t] = __builtin_amdgcn_wmma_f32_16x16x32_f16(
          false, a, false, b, (short)0, acc[t], false, false);
    }
  }

  int nc = lane & 15;
#pragma unroll
  for (int r = 0; r < 8; ++r) {
    int io = ibase + r + hi * 8;            // C/D layout: M = r (+8 for upper lanes)
    float zi  = Z [(size_t)hd * NN + io];
    float s1d = s1[(size_t)hd * NN + io];
    float mid = m [(size_t)hd * NN + io];
    // diagonal weight, rounded through f16 exactly like the A operand was
    float wii = (float)(_Float16)expw(s1d + s2s[io], mid);
#pragma unroll
    for (int t = 0; t < NT; ++t) {
      // Wh value rounded through f16 exactly like the packed B operand
      float bv = (float)(_Float16)Wsrc[(size_t)io * out_stride + hd * (NT * 16) + t * 16 + nc];
      float val = (acc[t][r] - wii * bv) / zi;
      if (DO_ELU) val = val > 0.f ? val : (__expf(val) - 1.f);
      outp[(size_t)io * out_stride + hd * (NT * 16) + t * 16 + nc] = val;
    }
  }
}

// ---------------- output layer prep: Wh_out = hcat @ W_out ; s1o/s2o ----------------
__global__ __launch_bounds__(256) void k_outprep(const float* __restrict__ hcat,
                                                 const float* __restrict__ Wout,
                                                 const float* __restrict__ a1o,
                                                 const float* __restrict__ a2o,
                                                 float* __restrict__ Who,
                                                 float* __restrict__ s1o,
                                                 float* __restrict__ s2o) {
  int n = blockIdx.x * 256 + threadIdx.x;
  if (n >= NN) return;
  float o[TT];
#pragma unroll
  for (int c = 0; c < TT; ++c) o[c] = 0.f;
  const float* hr = hcat + (size_t)n * 256;
  for (int k = 0; k < 256; ++k) {
    float hv = hr[k];
    const float* wr = Wout + (size_t)k * TT;
#pragma unroll
    for (int c = 0; c < TT; ++c) o[c] += hv * wr[c];
  }
  float d1 = 0.f, d2 = 0.f;
#pragma unroll
  for (int c = 0; c < TT; ++c) {
    Who[(size_t)n * TT + c] = o[c];
    d1 += o[c] * a1o[c];
    d2 += o[c] * a2o[c];
  }
  s1o[n] = d1; s2o[n] = d2;
}

extern "C" void kernel_launch(void* const* d_in, const int* in_sizes, int n_in,
                              void* d_out, int out_size, void* d_ws, size_t ws_size,
                              hipStream_t stream) {
  const float* x     = (const float*)d_in[0];
  // d_in[1] adj, d_in[2] adj_hat: unused by the reference module
  const float* Whds  = (const float*)d_in[3];
  const float* a1h   = (const float*)d_in[4];
  const float* a2h   = (const float*)d_in[5];
  const float* Wout  = (const float*)d_in[6];
  const float* a1o   = (const float*)d_in[7];
  const float* a2o   = (const float*)d_in[8];
  float* out = (float*)d_out;

  char* ws = (char*)d_ws;
  float*    h32   = (float*)(ws + OFF_H);
  float*    Wh    = (float*)(ws + OFF_WH);
  float*    s1    = (float*)(ws + OFF_S1);
  float*    s2    = (float*)(ws + OFF_S2);
  float*    topf  = (float*)(ws + OFF_TOPF);
  int*      topi  = (int*)  (ws + OFF_TOPI);
  float*    mrow  = (float*)(ws + OFF_M);
  float*    Zrow  = (float*)(ws + OFF_Z);
  _Float16* blobH = (_Float16*)(ws + OFF_BH);
  float*    hcat  = (float*)(ws + OFF_HCAT);
  float*    Who   = (float*)(ws + OFF_WHO);
  float*    s1o   = (float*)(ws + OFF_S1O);
  float*    s2o   = (float*)(ws + OFF_S2O);
  float*    topfo = (float*)(ws + OFF_TOPFO);
  int*      topio = (int*)  (ws + OFF_TOPIO);
  float*    mo    = (float*)(ws + OFF_MO);
  float*    Zo    = (float*)(ws + OFF_ZO);
  _Float16* blobO = (_Float16*)(ws + OFF_BO);

  k_prep_h <<<(NN * 64) / 256, 256, 0, stream>>>(x, h32);
  k_gemm_wh<<<(NN * 256) / 256, 256, 0, stream>>>(h32, Whds, Wh);
  k_s12    <<<(NHD * NN) / 256, 256, 0, stream>>>(Wh, a1h, a2h, s1, s2);
  k_top2   <<<NHD, 256, 0, stream>>>(s2, topf, topi);
  k_passA  <<<dim3(NN / 256, NHD), 256, 0, stream>>>(s1, s2, topf, topi, mrow, Zrow);
  k_packB  <<<(16 * 128 * 512) / 256, 256, 0, stream>>>(Wh, blobH, 16, 256);
  k_passB<4, true><<<dim3(NN / 128, NHD), 256, 0, stream>>>(s1, s2, mrow, Zrow,
                                                            blobH, Wh, hcat, 256);
  k_outprep<<<NN / 256, 256, 0, stream>>>(hcat, Wout, a1o, a2o, Who, s1o, s2o);
  k_top2   <<<1, 256, 0, stream>>>(s2o, topfo, topio);
  k_passA  <<<dim3(NN / 256, 1), 256, 0, stream>>>(s1o, s2o, topfo, topio, mo, Zo);
  k_packB  <<<(128 * 512) / 256, 256, 0, stream>>>(Who, blobO, 1, TT);
  k_passB<1, false><<<dim3(NN / 128, 1), 256, 0, stream>>>(s1o, s2o, mo, Zo,
                                                           blobO, Who, out, TT);
  (void)in_sizes; (void)n_in; (void)out_size; (void)ws_size;
}
